// MainAttention_58471684768049
// MI455X (gfx1250) — matrix operand order repaired and checked by reading, compiled, not run on hardware
//
#include <hip/hip_runtime.h>

typedef __attribute__((ext_vector_type(16))) __bf16 v16bf;
typedef __attribute__((ext_vector_type(8)))  float  v8f;

#define B_    16
#define N_    577
#define C_    768
#define H_    12
#define HD_   64
#define NP_   608              // N padded to multiple of 32
#define M_    (B_*N_)          // 9232 (== 577 * 16, exact 16-row tiles)
#define C3_   (3*C_)           // 2304
#define KEEP_ 404              // ceil(0.7*576) with fp rounding
#define SCALE_ 0.125f          // hd^-0.5

// d_out element offsets (f32), concatenated in return order
#define OUT_IDX   (M_*C_)                           // 7,090,176
#define OUT_INDEX (OUT_IDX + B_*KEEP_)              // 7,096,640
#define OUT_CLS   (OUT_INDEX + (size_t)B_*KEEP_*C_) // 12,060,992

__device__ __forceinline__ unsigned short f2bf(float f) {
  union { float f; unsigned u; } c; c.f = f;
  unsigned u = c.u; u += 0x7fffu + ((u >> 16) & 1u);
  return (unsigned short)(u >> 16);
}
__device__ __forceinline__ float bf2f(unsigned short h) {
  union { unsigned u; float f; } c; c.u = ((unsigned)h) << 16; return c.f;
}
__device__ __forceinline__ float bflo(unsigned u) {
  union { unsigned x; float f; } c; c.x = u << 16; return c.f;
}
__device__ __forceinline__ float bfhi(unsigned u) {
  union { unsigned x; float f; } c; c.x = u & 0xffff0000u; return c.f;
}

union Frag { uint4 u4[2]; v16bf v; };

// ---------------------------------------------------------------- f32 -> bf16
__global__ void cvt_kernel(const float* __restrict__ x, const float* __restrict__ wqkv,
                           const float* __restrict__ wp,
                           unsigned short* __restrict__ Xb, unsigned short* __restrict__ Wq,
                           unsigned short* __restrict__ Wp) {
  int i = blockIdx.x * blockDim.x + threadIdx.x;
  const int nx = M_ * C_, nq = C3_ * C_, np = C_ * C_;
  if (i < nx)                 Xb[i] = f2bf(x[i]);
  else if (i < nx + nq)       Wq[i - nx] = f2bf(wqkv[i - nx]);
  else if (i < nx + nq + np)  Wp[i - nx - nq] = f2bf(wp[i - nx - nq]);
}

// ------------------------------------------------- QKV GEMM: Xb[9232,768] @ Wq^T
// 4 waves/block; each wave computes 16 rows x 64 cols (4 WMMAs / k-step, A frag
// reused 4x). Fully unrolled k-loop, fragments loaded ahead of the WMMA group so
// the scheduler can pipeline loads of step k+1 under WMMAs of step k.
__global__ __launch_bounds__(128) void gemm_qkv(const unsigned short* __restrict__ Xb,
    const unsigned short* __restrict__ Wq,
    unsigned short* __restrict__ Q, unsigned short* __restrict__ K,
    unsigned short* __restrict__ Vt) {
  const int lane = threadIdx.x & 31;
  const int wave = threadIdx.x >> 5;
  const int m0 = (blockIdx.x % (M_ / 16)) * 16;
  const int nbase = (blockIdx.x / (M_ / 16)) * 256 + wave * 64;   // 64-aligned
  const int rowA = m0 + (lane & 15);
  const int aK = (lane >= 16) ? 16 : 0;   // A: lane-half K split at 16
  const int cb = nbase + (lane & 15);
  const int bK = (lane >= 16) ? 16 : 0;   // B: lane-half K split at 16
  v8f acc[4] = {{}, {}, {}, {}};
  const unsigned short* apBase = Xb + (size_t)rowA * C_ + aK;
  const unsigned short* bpBase = Wq + (size_t)cb * C_ + bK;
  __builtin_prefetch(apBase, 0, 1);
  __builtin_prefetch(bpBase, 0, 1);
#pragma unroll
  for (int k0 = 0; k0 < C_; k0 += 32) {
    Frag fa, fb0, fb1, fb2, fb3;
    fa.u4[0]  = *(const uint4*)(apBase + k0);
    fa.u4[1]  = *(const uint4*)(apBase + k0 + 8);
    fb0.u4[0] = *(const uint4*)(bpBase + 0 * 16 * C_ + k0);
    fb0.u4[1] = *(const uint4*)(bpBase + 0 * 16 * C_ + k0 + 8);
    fb1.u4[0] = *(const uint4*)(bpBase + 1 * 16 * C_ + k0);
    fb1.u4[1] = *(const uint4*)(bpBase + 1 * 16 * C_ + k0 + 8);
    fb2.u4[0] = *(const uint4*)(bpBase + 2 * 16 * C_ + k0);
    fb2.u4[1] = *(const uint4*)(bpBase + 2 * 16 * C_ + k0 + 8);
    fb3.u4[0] = *(const uint4*)(bpBase + 3 * 16 * C_ + k0);
    fb3.u4[1] = *(const uint4*)(bpBase + 3 * 16 * C_ + k0 + 8);
    acc[0] = __builtin_amdgcn_wmma_f32_16x16x32_bf16(false, fa.v, false, fb0.v,
                                                     (short)0, acc[0], false, false);
    acc[1] = __builtin_amdgcn_wmma_f32_16x16x32_bf16(false, fa.v, false, fb1.v,
                                                     (short)0, acc[1], false, false);
    acc[2] = __builtin_amdgcn_wmma_f32_16x16x32_bf16(false, fa.v, false, fb2.v,
                                                     (short)0, acc[2], false, false);
    acc[3] = __builtin_amdgcn_wmma_f32_16x16x32_bf16(false, fa.v, false, fb3.v,
                                                     (short)0, acc[3], false, false);
  }
  const int part = nbase / C_;            // uniform per wave: 0=q 1=k 2=v
  const int h = (nbase % C_) / HD_;       // uniform per wave
  const int rAdd = (lane >= 16) ? 8 : 0;
  const int dl = lane & 15;
#pragma unroll
  for (int g = 0; g < 8; ++g) {
    int row = m0 + g + rAdd;
    int bb = row / N_, n = row % N_;
    if (part == 0) {
      size_t base = ((size_t)(bb * H_ + h) * NP_ + n) * HD_;
#pragma unroll
      for (int c = 0; c < 4; ++c) Q[base + c * 16 + dl] = f2bf(acc[c][g]);
    } else if (part == 1) {
      size_t base = ((size_t)(bb * H_ + h) * NP_ + n) * HD_;
#pragma unroll
      for (int c = 0; c < 4; ++c) K[base + c * 16 + dl] = f2bf(acc[c][g]);
    } else {
      size_t base = ((size_t)(bb * H_ + h) * HD_) * NP_ + n;
#pragma unroll
      for (int c = 0; c < 4; ++c) Vt[base + (size_t)(c * 16 + dl) * NP_] = f2bf(acc[c][g]);
    }
  }
}

// ----------------------------------------------------------- flash attention
// 1 block = (b,h, 64-query-row tile); 1 wave = 16 rows; 32 keys per iteration.
// K/V tiles shared by all 4 waves: double-buffered async staging into LDS via
// global_load_async_to_lds_b128 — tile j+1 streams in while tile j is computed.
#define RT_ 10   // ceil(577/64)
__global__ __launch_bounds__(128) void attn_kernel(const unsigned short* __restrict__ Q,
    const unsigned short* __restrict__ K, const unsigned short* __restrict__ Vt,
    unsigned short* __restrict__ AO) {
  __shared__ __align__(16) unsigned short LDSK[2][32 * HD_];   // 32 keys x 64 d
  __shared__ __align__(16) unsigned short LDSV[2][HD_ * 32];   // 64 d x 32 keys
  __shared__ __align__(16) unsigned short P[4][16 * 32];       // per-wave P staging
  const int lane = threadIdx.x & 31;
  const int wave = threadIdx.x >> 5;
  const int rt = blockIdx.x % RT_;
  const int bh = blockIdx.x / RT_;
  const int bb = bh / H_, hh = bh % H_;
  const unsigned short* Qp = Q  + (size_t)bh * NP_ * HD_;
  const unsigned short* Kp = K  + (size_t)bh * NP_ * HD_;
  const unsigned short* Vp = Vt + (size_t)bh * HD_ * NP_;
  const int q0 = rt * 64 + wave * 16;
  const int myrow = q0 + (lane & 15);
  const int qr = (myrow < N_) ? myrow : (N_ - 1);   // clamp (results masked on store)

  // async-stage one 32-key K tile + V tile into the given LDS buffers
  auto stage = [&](int jj, unsigned short* ldk, unsigned short* ldv) {
    const int t = threadIdx.x;
#pragma unroll
    for (int cc = 0; cc < 2; ++cc) {
      int ch = t + cc * 128;                 // 256 x 16B chunks per tile
      int krow = ch >> 3;                    // key row 0..31
      int koffS = (ch & 7) * 8;              // shorts within row
      int key = jj + krow; if (key > N_ - 1) key = N_ - 1;
      unsigned kvoff = (unsigned)((key * HD_ + koffS) * 2);
      unsigned klds  = (unsigned)(size_t)(ldk + ch * 8);
      asm volatile("global_load_async_to_lds_b128 %0, %1, %2"
                   :: "v"(klds), "v"(kvoff), "s"(Kp) : "memory");
      int vrow = ch >> 2;                    // d row 0..63
      int voffS = (ch & 3) * 8;
      unsigned vvoff = (unsigned)((vrow * NP_ + jj + voffS) * 2);
      unsigned vlds  = (unsigned)(size_t)(ldv + ch * 8);
      asm volatile("global_load_async_to_lds_b128 %0, %1, %2"
                   :: "v"(vlds), "v"(vvoff), "s"(Vp) : "memory");
    }
  };

  Frag qa[2];                                        // Q rows, d-chunks 0 / 32
  {
    const int koff = (lane >= 16) ? 16 : 0;
#pragma unroll
    for (int cc = 0; cc < 2; ++cc) {
      const unsigned short* qp = Qp + (size_t)qr * HD_ + cc * 32 + koff;
      qa[cc].u4[0] = *(const uint4*)(qp);
      qa[cc].u4[1] = *(const uint4*)(qp + 8);
    }
  }
  v8f acc0 = {}, acc1 = {}, acc2 = {}, acc3 = {};
  float mrow[8], lrow[8];
#pragma unroll
  for (int g = 0; g < 8; ++g) { mrow[g] = -1e30f; lrow[g] = 0.f; }

  stage(0, LDSK[0], LDSV[0]);                        // prologue

  for (int j = 0; j < NP_; j += 32) {
    const int cur = (j >> 5) & 1;
    asm volatile("s_wait_asynccnt 0x0" ::: "memory"); // tile j resident (own wave)
    __syncthreads();                                  // ... and for all waves
    if (j + 32 < NP_) stage(j + 32, LDSK[cur ^ 1], LDSV[cur ^ 1]);
    const unsigned short* LK = LDSK[cur];
    const unsigned short* LV = LDSV[cur];

    // ---- scores: Q(16x64) @ K^T(64x16) x2 key groups ----
    v8f s[2];
    {
      const int koff = (lane >= 16) ? 16 : 0;
#pragma unroll
      for (int kg = 0; kg < 2; ++kg) {
        const unsigned short* kp = LK + (kg * 16 + (lane & 15)) * HD_;
        Frag kb0, kb1;
        kb0.u4[0] = *(const uint4*)(kp + koff);
        kb0.u4[1] = *(const uint4*)(kp + koff + 8);
        kb1.u4[0] = *(const uint4*)(kp + 32 + koff);
        kb1.u4[1] = *(const uint4*)(kp + 32 + koff + 8);
        v8f z = {};
        z = __builtin_amdgcn_wmma_f32_16x16x32_bf16(false, qa[0].v, false, kb0.v,
                                                    (short)0, z, false, false);
        z = __builtin_amdgcn_wmma_f32_16x16x32_bf16(false, qa[1].v, false, kb1.v,
                                                    (short)0, z, false, false);
        s[kg] = z;
      }
    }
    {
      int keyb = j + (lane & 15);
#pragma unroll
      for (int kg = 0; kg < 2; ++kg) {
        float msk = ((keyb + kg * 16) < N_) ? 0.f : -1e30f;
#pragma unroll
        for (int g = 0; g < 8; ++g) s[kg][g] = s[kg][g] * SCALE_ + msk;
      }
    }
    float p0[8], p1[8];
#pragma unroll
    for (int g = 0; g < 8; ++g) {                    // online softmax per row
      float tm = fmaxf(s[0][g], s[1][g]);
      tm = fmaxf(tm, __shfl_xor(tm, 1));
      tm = fmaxf(tm, __shfl_xor(tm, 2));
      tm = fmaxf(tm, __shfl_xor(tm, 4));
      tm = fmaxf(tm, __shfl_xor(tm, 8));
      float mnew = fmaxf(mrow[g], tm);
      float fac = __expf(mrow[g] - mnew);
      mrow[g] = mnew;
      float e0 = __expf(s[0][g] - mnew);
      float e1 = __expf(s[1][g] - mnew);
      float rs = e0 + e1;
      rs += __shfl_xor(rs, 1);
      rs += __shfl_xor(rs, 2);
      rs += __shfl_xor(rs, 4);
      rs += __shfl_xor(rs, 8);
      lrow[g] = lrow[g] * fac + rs;
      acc0[g] *= fac; acc1[g] *= fac; acc2[g] *= fac; acc3[g] *= fac;
      p0[g] = e0; p1[g] = e1;
    }
    // C-layout -> A-layout transpose of P through LDS (wave-synchronous)
    unsigned short* pl = P[wave];
    {
      int r = (lane >= 16) ? 8 : 0;
      int kc = lane & 15;
#pragma unroll
      for (int g = 0; g < 8; ++g) {
        pl[(r + g) * 32 + kc]      = f2bf(p0[g]);
        pl[(r + g) * 32 + 16 + kc] = f2bf(p1[g]);
      }
    }
    __builtin_amdgcn_wave_barrier();
    asm volatile("s_wait_dscnt 0" ::: "memory");
    Frag pa;
    {
      int r  = lane & 15;
      int kb = (lane >= 16) ? 16 : 0;
      pa.u4[0] = *(const uint4*)(pl + r * 32 + kb);
      pa.u4[1] = *(const uint4*)(pl + r * 32 + 8 + kb);
    }
    __builtin_amdgcn_wave_barrier();
    // ---- P(16x32) @ V(32x64) from LDS-staged V ----
    {
      const int koff = (lane >= 16) ? 16 : 0;
      const int dcol = lane & 15;
      Frag vb;
      const unsigned short* vp;
      vp = LV + (0 * 16 + dcol) * 32 + koff;
      vb.u4[0] = *(const uint4*)vp; vb.u4[1] = *(const uint4*)(vp + 8);
      acc0 = __builtin_amdgcn_wmma_f32_16x16x32_bf16(false, pa.v, false, vb.v,
                                                     (short)0, acc0, false, false);
      vp = LV + (1 * 16 + dcol) * 32 + koff;
      vb.u4[0] = *(const uint4*)vp; vb.u4[1] = *(const uint4*)(vp + 8);
      acc1 = __builtin_amdgcn_wmma_f32_16x16x32_bf16(false, pa.v, false, vb.v,
                                                     (short)0, acc1, false, false);
      vp = LV + (2 * 16 + dcol) * 32 + koff;
      vb.u4[0] = *(const uint4*)vp; vb.u4[1] = *(const uint4*)(vp + 8);
      acc2 = __builtin_amdgcn_wmma_f32_16x16x32_bf16(false, pa.v, false, vb.v,
                                                     (short)0, acc2, false, false);
      vp = LV + (3 * 16 + dcol) * 32 + koff;
      vb.u4[0] = *(const uint4*)vp; vb.u4[1] = *(const uint4*)(vp + 8);
      acc3 = __builtin_amdgcn_wmma_f32_16x16x32_bf16(false, pa.v, false, vb.v,
                                                     (short)0, acc3, false, false);
    }
    // next iteration's top-of-loop wait+barrier protects buffer reuse
  }
#pragma unroll
  for (int g = 0; g < 8; ++g) {
    float inv = (lrow[g] > 0.f) ? 1.f / lrow[g] : 0.f;
    acc0[g] *= inv; acc1[g] *= inv; acc2[g] *= inv; acc3[g] *= inv;
  }
  const int rAdd = (lane >= 16) ? 8 : 0;
  const int dc = lane & 15;
#pragma unroll
  for (int g = 0; g < 8; ++g) {
    int row = q0 + g + rAdd;
    if (row < N_) {
      size_t base = ((size_t)(bb * N_ + row)) * C_ + hh * HD_;
      AO[base +  0 + dc] = f2bf(acc0[g]);
      AO[base + 16 + dc] = f2bf(acc1[g]);
      AO[base + 32 + dc] = f2bf(acc2[g]);
      AO[base + 48 + dc] = f2bf(acc3[g]);
    }
  }
}

// ----------------------------------------- exact CLS-row softmax, head-mean
__global__ __launch_bounds__(256) void cls_kernel(const unsigned short* __restrict__ Q,
    const unsigned short* __restrict__ K, float* __restrict__ cls,
    float* __restrict__ out_cls) {
  __shared__ float q0s[HD_];
  __shared__ float red[256];
  const int b = blockIdx.x, t = threadIdx.x;
  float accp[3] = {0.f, 0.f, 0.f};
  for (int h = 0; h < H_; ++h) {
    const unsigned short* Qp = Q + ((size_t)(b * H_ + h)) * NP_ * HD_;
    const unsigned short* Kp = K + ((size_t)(b * H_ + h)) * NP_ * HD_;
    __syncthreads();
    if (t < HD_) q0s[t] = bf2f(Qp[t]);          // q row 0
    __syncthreads();
    float sv[3];
#pragma unroll
    for (int i = 0; i < 3; ++i) {
      int m = t + i * 256;
      if (m < N_) {
        const uint4* kp4 = (const uint4*)(Kp + (size_t)m * HD_);
        float s = 0.f;
#pragma unroll
        for (int c = 0; c < 8; ++c) {
          uint4 u = kp4[c];
          const float* qv = q0s + c * 8;
          s += bflo(u.x) * qv[0] + bfhi(u.x) * qv[1];
          s += bflo(u.y) * qv[2] + bfhi(u.y) * qv[3];
          s += bflo(u.z) * qv[4] + bfhi(u.z) * qv[5];
          s += bflo(u.w) * qv[6] + bfhi(u.w) * qv[7];
        }
        sv[i] = s * SCALE_;
      } else sv[i] = -1e30f;
    }
    float lm = fmaxf(fmaxf(sv[0], sv[1]), sv[2]);
    red[t] = lm; __syncthreads();
    for (int off = 128; off > 0; off >>= 1) { if (t < off) red[t] = fmaxf(red[t], red[t + off]); __syncthreads(); }
    float gm = red[0]; __syncthreads();
    float le = 0.f;
#pragma unroll
    for (int i = 0; i < 3; ++i) { sv[i] = __expf(sv[i] - gm); le += sv[i]; }
    red[t] = le; __syncthreads();
    for (int off = 128; off > 0; off >>= 1) { if (t < off) red[t] += red[t + off]; __syncthreads(); }
    float inv = 1.f / red[0]; __syncthreads();
#pragma unroll
    for (int i = 0; i < 3; ++i) accp[i] += sv[i] * inv;
  }
#pragma unroll
  for (int i = 0; i < 3; ++i) {
    int m = t + i * 256;
    if (m >= 1 && m < N_) {
      float v = accp[i] * (1.f / H_);
      cls[b * (N_ - 1) + (m - 1)] = v;
      out_cls[b * (N_ - 1) + (m - 1)] = v;
    }
  }
}

// ------------------------------------------------- out proj: AO @ Wp^T + bias
__global__ __launch_bounds__(128) void gemm_proj(const unsigned short* __restrict__ AO,
    const unsigned short* __restrict__ Wp, const float* __restrict__ bias,
    float* __restrict__ out) {
  const int lane = threadIdx.x & 31;
  const int wave = threadIdx.x >> 5;
  const int m0 = (blockIdx.x % (M_ / 16)) * 16;
  const int nbase = (blockIdx.x / (M_ / 16)) * 256 + wave * 64;
  const int rowA = m0 + (lane & 15);
  const int aK = (lane >= 16) ? 16 : 0;
  const int cb = nbase + (lane & 15);
  const int bK = (lane >= 16) ? 16 : 0;
  v8f acc[4] = {{}, {}, {}, {}};
  const unsigned short* apBase = AO + (size_t)rowA * C_ + aK;
  const unsigned short* bpBase = Wp + (size_t)cb * C_ + bK;
  __builtin_prefetch(apBase, 0, 1);
  __builtin_prefetch(bpBase, 0, 1);
#pragma unroll
  for (int k0 = 0; k0 < C_; k0 += 32) {
    Frag fa, fb0, fb1, fb2, fb3;
    fa.u4[0]  = *(const uint4*)(apBase + k0);
    fa.u4[1]  = *(const uint4*)(apBase + k0 + 8);
    fb0.u4[0] = *(const uint4*)(bpBase + 0 * 16 * C_ + k0);
    fb0.u4[1] = *(const uint4*)(bpBase + 0 * 16 * C_ + k0 + 8);
    fb1.u4[0] = *(const uint4*)(bpBase + 1 * 16 * C_ + k0);
    fb1.u4[1] = *(const uint4*)(bpBase + 1 * 16 * C_ + k0 + 8);
    fb2.u4[0] = *(const uint4*)(bpBase + 2 * 16 * C_ + k0);
    fb2.u4[1] = *(const uint4*)(bpBase + 2 * 16 * C_ + k0 + 8);
    fb3.u4[0] = *(const uint4*)(bpBase + 3 * 16 * C_ + k0);
    fb3.u4[1] = *(const uint4*)(bpBase + 3 * 16 * C_ + k0 + 8);
    acc[0] = __builtin_amdgcn_wmma_f32_16x16x32_bf16(false, fa.v, false, fb0.v,
                                                     (short)0, acc[0], false, false);
    acc[1] = __builtin_amdgcn_wmma_f32_16x16x32_bf16(false, fa.v, false, fb1.v,
                                                     (short)0, acc[1], false, false);
    acc[2] = __builtin_amdgcn_wmma_f32_16x16x32_bf16(false, fa.v, false, fb2.v,
                                                     (short)0, acc[2], false, false);
    acc[3] = __builtin_amdgcn_wmma_f32_16x16x32_bf16(false, fa.v, false, fb3.v,
                                                     (short)0, acc[3], false, false);
  }
  const int rAdd = (lane >= 16) ? 8 : 0;
  float bv[4];
#pragma unroll
  for (int c = 0; c < 4; ++c) bv[c] = bias[cb + c * 16];
#pragma unroll
  for (int g = 0; g < 8; ++g) {
    int row = m0 + g + rAdd;
#pragma unroll
    for (int c = 0; c < 4; ++c)
      out[(size_t)row * C_ + cb + c * 16] = acc[c][g] + bv[c];
  }
}

// --------------------------------- stable descending top-k via rank counting
__global__ void topk_kernel(const float* __restrict__ cls, float* __restrict__ idx_out) {
  const int b = blockIdx.x, t = threadIdx.x;
  if (t >= N_ - 1) return;
  const float v = cls[b * (N_ - 1) + t];
  int rank = 0;
  for (int j = 0; j < N_ - 1; ++j) {
    float vj = cls[b * (N_ - 1) + j];
    rank += (vj > v) || (vj == v && j < t);
  }
  if (rank < KEEP_) idx_out[b * KEEP_ + rank] = (float)t;
}

__global__ void idxbc_kernel(const float* __restrict__ idx_in, float* __restrict__ index_out) {
  size_t i = (size_t)blockIdx.x * blockDim.x + threadIdx.x;
  if (i >= (size_t)B_ * KEEP_ * C_) return;
  index_out[i] = idx_in[i / C_];
}

extern "C" void kernel_launch(void* const* d_in, const int* in_sizes, int n_in,
                              void* d_out, int out_size, void* d_ws, size_t ws_size,
                              hipStream_t stream) {
  const float* x      = (const float*)d_in[0];
  const float* qkv_w  = (const float*)d_in[1];
  const float* proj_w = (const float*)d_in[2];
  const float* proj_b = (const float*)d_in[3];
  float* out = (float*)d_out;

  char* ws = (char*)d_ws;
  size_t off = 0;
  unsigned short* Xb = (unsigned short*)(ws + off); off += (size_t)M_ * C_ * 2;
  unsigned short* Wq = (unsigned short*)(ws + off); off += (size_t)C3_ * C_ * 2;
  unsigned short* Wp = (unsigned short*)(ws + off); off += (size_t)C_ * C_ * 2;
  unsigned short* Qh = (unsigned short*)(ws + off); off += (size_t)B_ * H_ * NP_ * HD_ * 2;
  unsigned short* Kh = (unsigned short*)(ws + off); off += (size_t)B_ * H_ * NP_ * HD_ * 2;
  unsigned short* Vt = (unsigned short*)(ws + off); off += (size_t)B_ * H_ * HD_ * NP_ * 2;
  unsigned short* AO = (unsigned short*)(ws + off); off += (size_t)M_ * C_ * 2;
  float* clsws = (float*)(ws + off);                off += (size_t)B_ * (N_ - 1) * 4;

  const int total_cvt = M_ * C_ + C3_ * C_ + C_ * C_;
  cvt_kernel<<<(total_cvt + 255) / 256, 256, 0, stream>>>(x, qkv_w, proj_w, Xb, Wq, Wp);
  gemm_qkv<<<(M_ / 16) * (C3_ / 256), 128, 0, stream>>>(Xb, Wq, Qh, Kh, Vt);
  attn_kernel<<<B_ * H_ * RT_, 128, 0, stream>>>(Qh, Kh, Vt, AO);
  cls_kernel<<<B_, 256, 0, stream>>>(Qh, Kh, clsws, out + OUT_CLS);
  gemm_proj<<<(M_ / 16) * (C_ / 256), 128, 0, stream>>>(AO, Wp, proj_b, out);
  topk_kernel<<<B_, 576, 0, stream>>>(clsws, out + OUT_IDX);
  idxbc_kernel<<<(B_ * KEEP_ * C_ + 255) / 256, 256, 0, stream>>>(out + OUT_IDX, out + OUT_INDEX);
}
